// AdaptiveTransformerModel_6038724018439
// MI455X (gfx1250) — matrix operand order repaired and checked by reading, not compiled, not run on hardware
//
#include <hip/hip_runtime.h>
#include <hip/hip_bf16.h>

#define B_ 4
#define T_ 2048
#define D_ 1024
#define H_ 16
#define HD_ 64

typedef __bf16 bf16;
typedef bf16 v16bf __attribute__((ext_vector_type(16)));
typedef bf16 v8bf  __attribute__((ext_vector_type(8)));
typedef float v8f  __attribute__((ext_vector_type(8)));
typedef float v4f  __attribute__((ext_vector_type(4)));
typedef unsigned v4u __attribute__((ext_vector_type(4)));
typedef unsigned v8u __attribute__((ext_vector_type(8)));

__device__ __forceinline__ v16bf combine16(v8bf lo, v8bf hi) {
  return __builtin_shufflevector(lo, hi, 0, 1, 2, 3, 4, 5, 6, 7,
                                 8, 9, 10, 11, 12, 13, 14, 15);
}

// A-fragment (16x32 bf16): lane row = lane%16; elements 0..15 -> K = 2*kbase+0..15,
// kbase = 0 (lanes<16) or 8 (lanes>=16), i.e. K = 16*half + element.
__device__ __forceinline__ v16bf load_a_frag(const bf16* row, int kbase) {
  v8bf lo = *(const v8bf*)(row + 2 * kbase);
  v8bf hi = *(const v8bf*)(row + 2 * kbase + 8);
  return combine16(lo, hi);
}

// B-fragment (32x16 bf16): lane col = lane%16; lanes 0-15 hold K=0..15,
// lanes 16-31 hold K=16..31; p points at the 16 contiguous K values.
__device__ __forceinline__ v16bf load_b_frag(const bf16* p) {
  v8bf lo = *(const v8bf*)(p);
  v8bf hi = *(const v8bf*)(p + 8);
  return combine16(lo, hi);
}

__device__ __forceinline__ v8f wmma_bf16(v16bf a, v16bf b, v8f c) {
  return __builtin_amdgcn_wmma_f32_16x16x32_bf16(false, a, false, b, (short)0, c,
                                                 false, false);
}

// Async global->LDS copy of 16 bytes per lane (ASYNCcnt tracked, CDNA5).
__device__ __forceinline__ void async_copy_b128(void* lds_ptr, const void* gptr) {
  unsigned lds_addr = (unsigned)(size_t)lds_ptr;  // LDS aperture: low 32 bits
  asm volatile("global_load_async_to_lds_b128 %0, %1, off"
               :: "v"(lds_addr), "v"(gptr) : "memory");
}
__device__ __forceinline__ void wait_asynccnt0() {
  asm volatile("s_wait_asynccnt 0x0" ::: "memory");
}

// TDM: DMA a 32(rows) x 64(bf16) tile (row stride 64 elems) from global into
// LDS with an 8-half pad per row (pad_interval = 32 DWORDs, pad_amount = 4
// DWORDs) so the landed pitch is 72 halves — bank-conflict-free for B-frags.
__device__ __forceinline__ void tdm_load_k_tile(const bf16* gsrc, void* lds_dst,
                                                unsigned rows_total) {
  unsigned long long ga = (unsigned long long)(size_t)gsrc;
  v4u g0;
  g0[0] = 1u;                                   // count=1 valid descriptor
  g0[1] = (unsigned)(size_t)lds_dst;            // lds_addr
  g0[2] = (unsigned)(ga & 0xffffffffu);         // global_addr[31:0]
  g0[3] = (unsigned)((ga >> 32) & 0x01ffffffu)  // global_addr[56:32]
          | (2u << 30);                         // type = 2 ("image")
  v8u g1;
  g1[0] = (1u << 16)        // data_size = 2 bytes
        | (1u << 20)        // pad_enable
        | (4u << 22)        // pad_interval: 32 DWORDs (one 64-half row)
        | (3u << 25);       // pad_amount: 4 DWORDs (8 halves)
  g1[1] = (64u << 16);      // tensor_dim0 = 64 (bits 79:48, low half)
  g1[2] = (rows_total << 16); // tensor_dim1 low16 (bits 111:80)
  g1[3] = (64u << 16);      // tile_dim0 = 64 (bits 127:112)
  g1[4] = 32u;              // tile_dim1 = 32 (bits 143:128), tile_dim2 = 0
  g1[5] = 64u;              // tensor_dim0_stride = 64 (bits 207:160, low 32)
  g1[6] = 0u;
  g1[7] = 0u;
  asm volatile("tensor_load_to_lds %0, %1" :: "s"(g0), "s"(g1) : "memory");
}

// ---------------------------------------------------------------------------
// QKV projection: out[b,h,t,k] (bf16) = X[b,t,:] . W[h,:,k] + bias[h,k]
// Block tile 128M x 128N, K-step 32; wave tile 32M x 64N (8 WMMA / step).
// ---------------------------------------------------------------------------
__global__ __launch_bounds__(256)
void qkv_gemm_kernel(const float* __restrict__ X, const float* __restrict__ W,
                     const float* __restrict__ bias, bf16* __restrict__ out) {
  __shared__ __align__(16) bf16 xs[128][40];   // A tile [m][kk]
  __shared__ __align__(16) bf16 wt[128][40];   // B tile transposed [n][kk]
  const int tid  = threadIdx.x;
  const int lane = tid & 31;
  const int wv   = tid >> 5;
  const int wm   = wv & 3;     // 4 M-subtiles of 32 rows
  const int wn   = wv >> 2;    // 2 N-subtiles of 64 cols
  const int half = lane >> 4;
  const int l16  = lane & 15;
  const int m0   = blockIdx.x * 128;
  const int n0   = blockIdx.y * 128;

  const int we0 = tid * 16;                     // W staging: 16 floats along n
  const int wkk = we0 >> 7, wn_ = we0 & 127;
  const int wng = n0 + wn_;
  const int wh = wng >> 6, wk = wng & 63;

  v8f acc[2][4];
#pragma unroll
  for (int a = 0; a < 2; ++a)
#pragma unroll
    for (int f = 0; f < 4; ++f) acc[a][f] = {};

  for (int k0 = 0; k0 < D_; k0 += 32) {
    __syncthreads();
#pragma unroll
    for (int i = 0; i < 2; ++i) {  // stage X 128x32: 2 chunks of 8 per thread
      int c = tid + i * 256;
      int m = c >> 2, kb = (c & 3) * 8;
      const float* src = &X[(size_t)(m0 + m) * D_ + k0 + kb];
      v4f x0 = *(const v4f*)(src);
      v4f x1 = *(const v4f*)(src + 4);
      v8bf xp;
#pragma unroll
      for (int j = 0; j < 4; ++j) { xp[j] = (bf16)x0[j]; xp[j + 4] = (bf16)x1[j]; }
      *(v8bf*)&xs[m][kb] = xp;
      if (k0 + 32 < D_) __builtin_prefetch(src + 32, 0, 3);
    }
    {  // stage W 32x128 transposed: 16 contiguous floats -> 16 b16 scatters
      const float* src =
          &W[(size_t)wh * (D_ * HD_) + (size_t)(k0 + wkk) * HD_ + wk];
      v4f w0 = *(const v4f*)(src);
      v4f w1 = *(const v4f*)(src + 4);
      v4f w2 = *(const v4f*)(src + 8);
      v4f w3 = *(const v4f*)(src + 12);
#pragma unroll
      for (int j = 0; j < 4; ++j) {
        wt[wn_ + j][wkk]      = (bf16)w0[j];
        wt[wn_ + 4 + j][wkk]  = (bf16)w1[j];
        wt[wn_ + 8 + j][wkk]  = (bf16)w2[j];
        wt[wn_ + 12 + j][wkk] = (bf16)w3[j];
      }
      if (k0 + 32 < D_) __builtin_prefetch(src + 32 * HD_, 0, 3);
    }
    __syncthreads();

    // load all fragments, then batch the 8 WMMAs
    v16bf a0 = load_a_frag(&xs[wm * 32 + l16][0], half ? 8 : 0);
    v16bf a1 = load_a_frag(&xs[wm * 32 + 16 + l16][0], half ? 8 : 0);
    v16bf bfr[4];
#pragma unroll
    for (int f = 0; f < 4; ++f)
      bfr[f] = load_b_frag(&wt[wn * 64 + f * 16 + l16][half ? 16 : 0]);
#pragma unroll
    for (int f = 0; f < 4; ++f) {
      acc[0][f] = wmma_bf16(a0, bfr[f], acc[0][f]);
      acc[1][f] = wmma_bf16(a1, bfr[f], acc[1][f]);
    }
  }
#pragma unroll
  for (int a = 0; a < 2; ++a)
#pragma unroll
    for (int f = 0; f < 4; ++f) {
      int ng = n0 + wn * 64 + f * 16 + l16;
      int h = ng >> 6, k = ng & 63;
      float bv = bias[h * HD_ + k];
#pragma unroll
      for (int r = 0; r < 8; ++r) {
        int mg = m0 + wm * 32 + a * 16 + r + 8 * half;  // row = b*T + t
        int bb = mg >> 11, t = mg & (T_ - 1);
        out[(((size_t)(bb * H_ + h)) * T_ + t) * HD_ + k] =
            (bf16)(acc[a][f][r] + bv);
      }
    }
}

// ---------------------------------------------------------------------------
// Flash attention: per (b,h), 128 rows/block, 8 waves x 16 rows, 32-key steps.
// K tile staged via Tensor Data Mover; V staged transposed via registers.
// ---------------------------------------------------------------------------
__global__ __launch_bounds__(256)
void flash_attn_kernel(const bf16* __restrict__ Q, const bf16* __restrict__ K,
                       const bf16* __restrict__ V, bf16* __restrict__ A2,
                       float* __restrict__ normsum) {
  __shared__ __align__(16) bf16 ks[32][72];       // K tile [key][comp] (TDM pad)
  __shared__ __align__(16) bf16 vt[64][40];       // V tile transposed [comp][key]
  __shared__ __align__(16) bf16 pl[8][16][40];    // per-wave P staging

  const int tid  = threadIdx.x;
  const int lane = tid & 31;
  const int wv   = tid >> 5;
  const int half = lane >> 4;
  const int l16  = lane & 15;
  const int b    = blockIdx.z;
  const int h    = blockIdx.y;
  const int q0   = blockIdx.x * 128 + wv * 16;
  const size_t base = ((size_t)(b * H_ + h)) * T_ * HD_;
  const bf16* Qb = Q + base;
  const bf16* Kb = K + base;
  const bf16* Vb = V + base;

  const int ss = tid >> 3, sc = (tid & 7) * 8;    // V staging coords

  // Q fragments with softmax scale 1/sqrt(64)=0.125 folded in (exact in bf16)
  v16bf aq0, aq1;
  {
    const bf16* qrow = Qb + (size_t)(q0 + l16) * HD_;
    v16bf t0 = load_a_frag(qrow, half ? 8 : 0);
    v16bf t1 = load_a_frag(qrow + 32, half ? 8 : 0);
#pragma unroll
    for (int i = 0; i < 16; ++i) {
      aq0[i] = (bf16)((float)t0[i] * 0.125f);
      aq1[i] = (bf16)((float)t1[i] * 0.125f);
    }
  }

  float mr[8], lr[8];
  v8f acc[4];
#pragma unroll
  for (int r = 0; r < 8; ++r) { mr[r] = -1e30f; lr[r] = 0.f; }
#pragma unroll
  for (int f = 0; f < 4; ++f) acc[f] = {};

  for (int s0 = 0; s0 < T_; s0 += 32) {
    __syncthreads();
    if (wv == 0)  // one wave launches the K-tile DMA
      tdm_load_k_tile(Kb + (size_t)s0 * HD_, &ks[0][0], T_);
    {  // V tile: register round-trip, transposed scatter into LDS
      v8bf vvv = *(const v8bf*)(Vb + (size_t)(s0 + ss) * HD_ + sc);
#pragma unroll
      for (int j = 0; j < 8; ++j) vt[sc + j][ss] = vvv[j];
    }
    if (s0 + 32 < T_) {
      __builtin_prefetch(Kb + (size_t)(s0 + 32 + ss) * HD_ + sc, 0, 3);
      __builtin_prefetch(Vb + (size_t)(s0 + 32 + ss) * HD_ + sc, 0, 3);
    }
    if (wv == 0) __builtin_amdgcn_s_wait_tensorcnt(0);
    __syncthreads();

    // S tiles: keys 0..15 (sa0) and 16..31 (sa1); load frags, then WMMA batch
    v16bf bk00 = load_b_frag(&ks[l16][(half ? 16 : 0)]);
    v16bf bk01 = load_b_frag(&ks[16 + l16][(half ? 16 : 0)]);
    v16bf bk10 = load_b_frag(&ks[l16][32 + (half ? 16 : 0)]);
    v16bf bk11 = load_b_frag(&ks[16 + l16][32 + (half ? 16 : 0)]);
    v8f sa0 = {}, sa1 = {};
    sa0 = wmma_bf16(aq0, bk00, sa0);
    sa1 = wmma_bf16(aq0, bk01, sa1);
    sa0 = wmma_bf16(aq1, bk10, sa0);
    sa1 = wmma_bf16(aq1, bk11, sa1);

    // streaming softmax on D-layout (row = vgpr, col = lane within half-wave)
#pragma unroll
    for (int r = 0; r < 8; ++r) {
      float v0 = sa0[r], v1 = sa1[r];
      float mx = fmaxf(v0, v1);
#pragma unroll
      for (int msk = 1; msk < 16; msk <<= 1)
        mx = fmaxf(mx, __shfl_xor(mx, msk, 32));
      float mn = fmaxf(mr[r], mx);
      float corr = __expf(mr[r] - mn);
      float p0 = __expf(v0 - mn);
      float p1 = __expf(v1 - mn);
      float ls = p0 + p1;
#pragma unroll
      for (int msk = 1; msk < 16; msk <<= 1)
        ls += __shfl_xor(ls, msk, 32);
      lr[r] = lr[r] * corr + ls;
      mr[r] = mn;
#pragma unroll
      for (int f = 0; f < 4; ++f) acc[f][r] *= corr;
      int row = r + 8 * half;
      pl[wv][row][l16] = (bf16)p0;
      pl[wv][row][16 + l16] = (bf16)p1;
    }

    // P (16x32) x V (32x64)
    v16bf ap = load_a_frag(&pl[wv][l16][0], half ? 8 : 0);
    v16bf bvf[4];
#pragma unroll
    for (int f = 0; f < 4; ++f)
      bvf[f] = load_b_frag(&vt[f * 16 + l16][half ? 16 : 0]);
#pragma unroll
    for (int f = 0; f < 4; ++f) acc[f] = wmma_bf16(ap, bvf[f], acc[f]);
  }

  // finalize: divide by softmax denom, write bf16, accumulate row L2 norms
#pragma unroll
  for (int r = 0; r < 8; ++r) {
    float inv = 1.f / lr[r];
    int tok = q0 + r + 8 * half;
    size_t arow = ((size_t)b * T_ + tok) * (size_t)(H_ * HD_);
    float ss2 = 0.f;
#pragma unroll
    for (int f = 0; f < 4; ++f) {
      float o = acc[f][r] * inv;
      ss2 += o * o;
      A2[arow + h * HD_ + f * 16 + l16] = (bf16)o;
    }
#pragma unroll
    for (int msk = 1; msk < 16; msk <<= 1)
      ss2 += __shfl_xor(ss2, msk, 32);
    if (l16 == 0) atomicAdd(&normsum[h], sqrtf(ss2));
  }
}

__global__ void zero16(float* p) {
  if (threadIdx.x < H_) p[threadIdx.x] = 0.f;
}

__global__ __launch_bounds__(256)
void prep_kernel(const float* __restrict__ normsum, const float* __restrict__ gate,
                 const float* __restrict__ bo, float* __restrict__ scale,
                 float* __restrict__ bias2) {
  int tid = threadIdx.x;
  if (tid < H_) {
    float mean = normsum[tid] / (float)(B_ * T_);
    float denom = fmaxf(mean, 1e-5f);
    float g = fminf(fmaxf(gate[tid], 0.f), 1.f);
    scale[tid] = g / ((float)H_ * denom);
  }
  for (int d = tid; d < D_; d += 256) {
    float a = 0.f;
#pragma unroll
    for (int hh = 0; hh < H_; ++hh) {
      float g = fminf(fmaxf(gate[hh], 0.f), 1.f);
      a += g * bo[hh * D_ + d];
    }
    bias2[d] = a / (float)H_;
  }
}

// ---------------------------------------------------------------------------
// Output projection: out[m][d] = sum_n A2[m][n] * (scale[n>>6]*Wo[n][d]) + b2[d]
// A2 tile staged by async global->LDS copies; scale folded into Wo staging.
// ---------------------------------------------------------------------------
__global__ __launch_bounds__(256)
void out_gemm_kernel(const bf16* __restrict__ A2, const float* __restrict__ Wo,
                     const float* __restrict__ scale,
                     const float* __restrict__ bias2, float* __restrict__ out) {
  __shared__ __align__(16) bf16 xs[128][40];
  __shared__ __align__(16) bf16 wt[128][40];
  const int tid  = threadIdx.x;
  const int lane = tid & 31;
  const int wv   = tid >> 5;
  const int wm   = wv & 3;
  const int wn   = wv >> 2;
  const int half = lane >> 4;
  const int l16  = lane & 15;
  const int m0   = blockIdx.x * 128;
  const int n0   = blockIdx.y * 128;

  const int we0 = tid * 16;
  const int wkk = we0 >> 7, wn_ = we0 & 127;

  v8f acc[2][4];
#pragma unroll
  for (int a = 0; a < 2; ++a)
#pragma unroll
    for (int f = 0; f < 4; ++f) acc[a][f] = {};

  for (int k0 = 0; k0 < D_; k0 += 32) {
    __syncthreads();
#pragma unroll
    for (int i = 0; i < 2; ++i) {  // stage A2 128x32: pure async copies
      int c = tid + i * 256;
      int m = c >> 2, kb = (c & 3) * 8;
      const bf16* src = &A2[(size_t)(m0 + m) * D_ + k0 + kb];
      async_copy_b128(&xs[m][kb], src);
      if (k0 + 32 < D_) __builtin_prefetch(src + 32, 0, 3);
    }
    {  // stage Wo transposed with per-head scale folded into rows (K-dim)
      float sc = scale[(k0 + wkk) >> 6];
      const float* src = &Wo[(size_t)(k0 + wkk) * D_ + n0 + wn_];
      v4f w0 = *(const v4f*)(src);
      v4f w1 = *(const v4f*)(src + 4);
      v4f w2 = *(const v4f*)(src + 8);
      v4f w3 = *(const v4f*)(src + 12);
#pragma unroll
      for (int j = 0; j < 4; ++j) {
        wt[wn_ + j][wkk]      = (bf16)(w0[j] * sc);
        wt[wn_ + 4 + j][wkk]  = (bf16)(w1[j] * sc);
        wt[wn_ + 8 + j][wkk]  = (bf16)(w2[j] * sc);
        wt[wn_ + 12 + j][wkk] = (bf16)(w3[j] * sc);
      }
      if (k0 + 32 < D_) __builtin_prefetch(src + 32 * D_, 0, 3);
    }
    wait_asynccnt0();
    __syncthreads();

    v16bf a0 = load_a_frag(&xs[wm * 32 + l16][0], half ? 8 : 0);
    v16bf a1 = load_a_frag(&xs[wm * 32 + 16 + l16][0], half ? 8 : 0);
    v16bf bfr[4];
#pragma unroll
    for (int f = 0; f < 4; ++f)
      bfr[f] = load_b_frag(&wt[wn * 64 + f * 16 + l16][half ? 16 : 0]);
#pragma unroll
    for (int f = 0; f < 4; ++f) {
      acc[0][f] = wmma_bf16(a0, bfr[f], acc[0][f]);
      acc[1][f] = wmma_bf16(a1, bfr[f], acc[1][f]);
    }
  }
#pragma unroll
  for (int a = 0; a < 2; ++a)
#pragma unroll
    for (int f = 0; f < 4; ++f) {
      int ng = n0 + wn * 64 + f * 16 + l16;
      float bv = bias2[ng];
#pragma unroll
      for (int r = 0; r < 8; ++r) {
        int mg = m0 + wm * 32 + a * 16 + r + 8 * half;
        out[(size_t)mg * D_ + ng] = acc[a][f][r] + bv;
      }
    }
}

extern "C" void kernel_launch(void* const* d_in, const int* in_sizes, int n_in,
                              void* d_out, int out_size, void* d_ws,
                              size_t ws_size, hipStream_t stream) {
  const float* hidden = (const float*)d_in[0];
  const float* Wq = (const float*)d_in[1];
  const float* bq = (const float*)d_in[2];
  const float* Wk = (const float*)d_in[3];
  const float* bk = (const float*)d_in[4];
  const float* Wv = (const float*)d_in[5];
  const float* bv = (const float*)d_in[6];
  const float* Wo = (const float*)d_in[7];
  const float* bo = (const float*)d_in[8];
  const float* gate = (const float*)d_in[9];
  float* out = (float*)d_out;

  char* ws = (char*)d_ws;
  const size_t qkv_bytes = (size_t)B_ * H_ * T_ * HD_ * sizeof(bf16);  // 16 MiB
  bf16* Qb = (bf16*)(ws);
  bf16* Kb = (bf16*)(ws + qkv_bytes);
  bf16* Vb = (bf16*)(ws + 2 * qkv_bytes);
  bf16* A2 = (bf16*)(ws + 3 * qkv_bytes);
  float* normsum = (float*)(ws + 4 * qkv_bytes);
  float* scale = normsum + H_;
  float* bias2 = scale + H_;

  zero16<<<dim3(1), dim3(32), 0, stream>>>(normsum);

  dim3 gg((B_ * T_) / 128, D_ / 128, 1);
  qkv_gemm_kernel<<<gg, 256, 0, stream>>>(hidden, Wq, bq, Qb);
  qkv_gemm_kernel<<<gg, 256, 0, stream>>>(hidden, Wk, bk, Kb);
  qkv_gemm_kernel<<<gg, 256, 0, stream>>>(hidden, Wv, bv, Vb);

  flash_attn_kernel<<<dim3(T_ / 128, H_, B_), 256, 0, stream>>>(Qb, Kb, Vb, A2,
                                                                normsum);
  prep_kernel<<<dim3(1), dim3(256), 0, stream>>>(normsum, gate, bo, scale, bias2);
  out_gemm_kernel<<<gg, 256, 0, stream>>>(A2, Wo, scale, bias2, out);
}
